// LinearAttention_2207613190677
// MI455X (gfx1250) — compile-verified, matrix operand order checked by reading, not run on hardware
//
#include <hip/hip_runtime.h>

// ---------------------------------------------------------------------------
// LinearAttention on MI455X (gfx1250, wave32, WMMA).
//
// Precision plan (roofline): 140 GFLOP / ~0.5GB traffic. bf16 WMMA
// (16x16x32 = 16384 FLOP/instr) balances matrix-pipe time against HBM time
// (23.3 TB/s -> ~21us). Convert operands to bf16 once, fp32 accumulate,
// fp32 softmax.
//
// Data movement: big GEMMs stage tiles in LDS via CDNA5 async DMA
// (GLOBAL_LOAD_ASYNC_TO_LDS_B128, ASYNCcnt-tracked, double buffered) so one
// copy per workgroup feeds all 8 waves through ds_load_b128. LDS tile rows
// are padded to 80B so the 16-lane b128 fragment reads land on banks
// 20*r mod 64 (gap >= 4 banks) -> conflict-free, vs 4-way conflicts at 64B.
// ---------------------------------------------------------------------------

typedef __attribute__((ext_vector_type(16))) __bf16 v16bf;
typedef __attribute__((ext_vector_type(8)))  float  v8f;

static __device__ __forceinline__ unsigned short f2bf(float f) {
    union { float f; unsigned u; } c; c.f = f;
    unsigned r = c.u + 0x7FFFu + ((c.u >> 16) & 1u);   // round-to-nearest-even
    return (unsigned short)(r >> 16);
}

// ---------------------------------------------------------------------------
// fp32 -> bf16 conversion (grid-stride)
// ---------------------------------------------------------------------------
__global__ void f32_to_bf16_kernel(const float* __restrict__ in,
                                   unsigned short* __restrict__ out,
                                   long long n) {
    long long i = (long long)blockIdx.x * blockDim.x + threadIdx.x;
    long long stride = (long long)gridDim.x * blockDim.x;
    for (; i < n; i += stride) out[i] = f2bf(in[i]);
}

// ---------------------------------------------------------------------------
// 16x32 bf16 fragment per ISA VGPR layout:
//   lanes 0-15  : row = lane,    K = {k0..k0+7, k0+16..k0+23}
//   lanes 16-31 : row = lane-16, K = {k0+8..k0+15, k0+24..k0+31}
// Operands are K-contiguous (NT GEMM) -> two 16B loads per lane.
// ---------------------------------------------------------------------------
static __device__ __forceinline__ v16bf load_frag_g(const unsigned short* __restrict__ p,
                                                    int row0, long long ld, int k0,
                                                    int r, int half) {
    const unsigned short* base = p + (long long)(row0 + r) * ld + k0 + half * 8;
    union { uint4 u[2]; v16bf v; } un;
    un.u[0] = *reinterpret_cast<const uint4*>(base);
    un.u[1] = *reinterpret_cast<const uint4*>(base + 16);
    return un.v;
}

// Padded LDS tile row stride: 32 K-elements (64B) + 16B pad = 40 elements.
constexpr int LROW = 40;

// Fragment sourced from an LDS tile laid out [row][LROW] (K-contiguous).
static __device__ __forceinline__ v16bf load_frag_lds(const unsigned short* tile,
                                                      int row, int r, int half) {
    const unsigned short* base = tile + (row + r) * LROW + half * 8;
    union { uint4 u[2]; v16bf v; } un;
    un.u[0] = *reinterpret_cast<const uint4*>(base);        // ds_load_b128
    un.u[1] = *reinterpret_cast<const uint4*>(base + 16);   // ds_load_b128
    return un.v;
}

// Async DMA of one 16-byte chunk global -> LDS (ASYNCcnt-tracked, no VGPR
// return traffic). No builtin on this toolchain -> inline CDNA5 asm.
static __device__ __forceinline__ void async_copy_b128(unsigned lds_byte_addr,
                                                       const void* gaddr) {
    asm volatile("global_load_async_to_lds_b128 %0, %1, off"
                 :: "v"(lds_byte_addr), "v"(gaddr) : "memory");
}
static __device__ __forceinline__ void wait_asynccnt0() {
    asm volatile("s_wait_asynccnt 0x0" ::: "memory");
}

// ---------------------------------------------------------------------------
// Generic batched NT GEMM: C[z] = A[z] (MxK) * B[z] (NxK)^T  (+bias)
// Per-batch offsets: off = (z/zdiv)*s1 + (z%zdiv)*s2 (covers b/h splits).
// BMxBN block tile; WR x WC waves; each wave (BM/WR)x(BN/WC) via 16x16 WMMA.
// STAGE=true: double-buffered, bank-conflict-padded LDS tiles filled by
// async DMA (BK=32). All problem dims divide the tiles exactly (no guards).
// ---------------------------------------------------------------------------
template<int BM, int BN, int WR, int WC, bool OUT_BF16, bool BIAS, bool STAGE>
__global__ __launch_bounds__(WR * WC * 32)
void gemm_nt_bf16(const unsigned short* __restrict__ A,
                  const unsigned short* __restrict__ B,
                  void* __restrict__ C,
                  const float* __restrict__ bias,
                  int K, long long lda, long long ldb, long long ldc,
                  int zdiv,
                  long long sA1, long long sA2,
                  long long sB1, long long sB2,
                  long long sC1, long long sC2) {
    constexpr int WTM = BM / WR, WTN = BN / WC;
    constexpr int MT = WTM / 16, NT = WTN / 16;
    constexpr int NTHR = WR * WC * 32;
    constexpr int ATILE = BM * LROW;               // elements per A buffer
    constexpr int BTILE = BN * LROW;
    constexpr int BUF   = ATILE + BTILE;

    __shared__ __align__(16) unsigned short smem[STAGE ? 2 * BUF : 8];

    const int lane = threadIdx.x & 31;
    const int wid  = threadIdx.x >> 5;
    const int wr   = wid / WC, wc = wid % WC;
    const int r    = lane & 15, half = lane >> 4;

    const int z = blockIdx.z;
    const unsigned short* Ab = A + (long long)(z / zdiv) * sA1 + (long long)(z % zdiv) * sA2;
    const unsigned short* Bb = B + (long long)(z / zdiv) * sB1 + (long long)(z % zdiv) * sB2;
    const long long offC = (long long)(z / zdiv) * sC1 + (long long)(z % zdiv) * sC2;

    const int blockRow = blockIdx.y * BM;
    const int blockCol = blockIdx.x * BN;
    const int rowBase  = blockRow + wr * WTM;
    const int colBase  = blockCol + wc * WTN;

    v8f acc[MT][NT];
#pragma unroll
    for (int i = 0; i < MT; ++i)
#pragma unroll
        for (int j = 0; j < NT; ++j)
            acc[i][j] = (v8f){0.f, 0.f, 0.f, 0.f, 0.f, 0.f, 0.f, 0.f};

    if constexpr (STAGE) {
        // Issue async DMA for one BK=32 tile pair into buffer p.
        // A tile: BM rows x 64B (+16B pad); B tile: BN rows x 64B (+16B pad).
        auto issue_tile = [&](int p, int k0) {
            const int t = threadIdx.x;
            unsigned base = (unsigned)(unsigned long long)&smem[p * BUF];
            constexpr int ACH = BM * 4;
#pragma unroll
            for (int c = t; c < ACH; c += NTHR) {
                const int row = c >> 2, seg = c & 3;
                async_copy_b128(base + row * (LROW * 2) + seg * 16,
                                Ab + (long long)(blockRow + row) * lda + k0 + seg * 8);
            }
            unsigned baseB = base + ATILE * 2;
            constexpr int BCH = BN * 4;
#pragma unroll
            for (int c = t; c < BCH; c += NTHR) {
                const int row = c >> 2, seg = c & 3;
                async_copy_b128(baseB + row * (LROW * 2) + seg * 16,
                                Bb + (long long)(blockCol + row) * ldb + k0 + seg * 8);
            }
        };

        issue_tile(0, 0);
        wait_asynccnt0();
        __syncthreads();

        int p = 0;
        for (int k0 = 0; k0 < K; k0 += 32) {
            if (k0 + 32 < K) issue_tile(p ^ 1, k0 + 32);   // prefetch next tile

            const unsigned short* tA = &smem[p * BUF];
            const unsigned short* tB = &smem[p * BUF + ATILE];
            v16bf af[MT], bfq[NT];
#pragma unroll
            for (int im = 0; im < MT; ++im)
                af[im] = load_frag_lds(tA, wr * WTM + im * 16, r, half);
#pragma unroll
            for (int in = 0; in < NT; ++in)
                bfq[in] = load_frag_lds(tB, wc * WTN + in * 16, r, half);
#pragma unroll
            for (int im = 0; im < MT; ++im)
#pragma unroll
                for (int in = 0; in < NT; ++in)
                    acc[im][in] = __builtin_amdgcn_wmma_f32_16x16x32_bf16(
                        false, af[im], false, bfq[in], (short)0, acc[im][in],
                        false, false);

            wait_asynccnt0();    // my prefetch DMAs landed
            __syncthreads();     // everyone done reading buffer p
            p ^= 1;
        }
    } else {
        for (int k0 = 0; k0 < K; k0 += 32) {
            v16bf af[MT], bfq[NT];
#pragma unroll
            for (int im = 0; im < MT; ++im)
                af[im] = load_frag_g(Ab, rowBase + im * 16, lda, k0, r, half);
#pragma unroll
            for (int in = 0; in < NT; ++in)
                bfq[in] = load_frag_g(Bb, colBase + in * 16, ldb, k0, r, half);
#pragma unroll
            for (int im = 0; im < MT; ++im)
#pragma unroll
                for (int in = 0; in < NT; ++in)
                    acc[im][in] = __builtin_amdgcn_wmma_f32_16x16x32_bf16(
                        false, af[im], false, bfq[in], (short)0, acc[im][in],
                        false, false);
        }
    }

    // C/D layout: VGPR j, lanes 0-15 -> (M = j,   N = lane)
    //                     lanes 16-31 -> (M = j+8, N = lane-16)
#pragma unroll
    for (int im = 0; im < MT; ++im)
#pragma unroll
        for (int in = 0; in < NT; ++in)
#pragma unroll
            for (int j = 0; j < 8; ++j) {
                const int row = rowBase + im * 16 + half * 8 + j;
                const int col = colBase + in * 16 + r;
                float v = acc[im][in][j];
                if (BIAS) v += bias[col];
                const long long ci = offC + (long long)row * ldc + col;
                if (OUT_BF16) ((unsigned short*)C)[ci] = f2bf(v);
                else          ((float*)C)[ci] = v;
            }
}

// ---------------------------------------------------------------------------
// q softmax over head_dim (64 contiguous elems). One wave per (row, head)
// group; lane handles cols {lane, lane+32}; shfl reductions (wave32).
// Output: bf16 q [16384 x 1024].
// ---------------------------------------------------------------------------
__global__ __launch_bounds__(256)
void softmax_q_kernel(const float* __restrict__ qkv, unsigned short* __restrict__ qb) {
    const int lane = threadIdx.x & 31;
    const int g = blockIdx.x * 8 + (threadIdx.x >> 5);     // group in [0, 16384*16)
    const int row = g >> 4, h = g & 15;
    const long long base = (long long)row * 3072 + h * 64;
    float x0 = qkv[base + lane];
    float x1 = qkv[base + lane + 32];
    float m = fmaxf(x0, x1);
#pragma unroll
    for (int off = 16; off; off >>= 1) m = fmaxf(m, __shfl_xor(m, off, 32));
    float e0 = __expf(x0 - m), e1 = __expf(x1 - m);
    float s = e0 + e1;
#pragma unroll
    for (int off = 16; off; off >>= 1) s += __shfl_xor(s, off, 32);
    const float inv = 1.f / s;
    const long long ob = (long long)row * 1024 + h * 64;
    qb[ob + lane]      = f2bf(e0 * inv);
    qb[ob + lane + 32] = f2bf(e1 * inv);
}

// ---------------------------------------------------------------------------
// k softmax over sequence (N=4096, column-wise per (b,h,d)). One block per
// (b,h): 256 threads = 64 cols x 4 row-stripes; LDS tree for column max/sum.
// Each (b,h) slab is 1MB fp32; all 64 slabs sit in the 192MB L2, so the
// three passes re-read from L2, not HBM. Output written TRANSPOSED as
// bf16 kT [(b*16+h)][d][n] so that k^T·v becomes an NT WMMA GEMM.
// ---------------------------------------------------------------------------
__global__ __launch_bounds__(256)
void softmax_k_kernel(const float* __restrict__ qkv, unsigned short* __restrict__ kt) {
    __shared__ float red[256];
    __shared__ float cmax[64];
    __shared__ float csum[64];
    const int z = blockIdx.x;                 // b*16 + h
    const int b = z >> 4, h = z & 15;
    const int t = threadIdx.x;
    const int d = t & 63, rq = t >> 6;
    const long long colBase = 1024 + (long long)h * 64 + d;

    float m = -INFINITY;
    for (int r = rq; r < 4096; r += 4)
        m = fmaxf(m, qkv[((long long)(b * 4096 + r)) * 3072 + colBase]);
    red[t] = m; __syncthreads();
    if (t < 64)
        cmax[t] = fmaxf(fmaxf(red[t], red[t + 64]), fmaxf(red[t + 128], red[t + 192]));
    __syncthreads();

    float s = 0.f;
    const float cm = cmax[d];
    for (int r = rq; r < 4096; r += 4)
        s += __expf(qkv[((long long)(b * 4096 + r)) * 3072 + colBase] - cm);
    red[t] = s; __syncthreads();
    if (t < 64)
        csum[t] = red[t] + red[t + 64] + red[t + 128] + red[t + 192];
    __syncthreads();

    const float inv = 1.f / csum[d];
    const long long ob = ((long long)z * 64 + d) * 4096;
    for (int r = rq; r < 4096; r += 4) {
        float v = __expf(qkv[((long long)(b * 4096 + r)) * 3072 + colBase] - cm) * inv;
        kt[ob + r] = f2bf(v);
    }
}

// ---------------------------------------------------------------------------
// v: fp32 (B,N,3C slice) -> bf16 transposed vT [(b*16+h)][e][n].
// Mapped so reads are contiguous (e fastest); scattered bf16 writes.
// ---------------------------------------------------------------------------
__global__ void transpose_v_kernel(const float* __restrict__ qkv,
                                   unsigned short* __restrict__ vt) {
    long long i = (long long)blockIdx.x * blockDim.x + threadIdx.x;
    const long long total = 64LL * 64 * 4096;
    const long long stride = (long long)gridDim.x * blockDim.x;
    for (; i < total; i += stride) {
        const int e = (int)(i & 63);
        const int n = (int)((i >> 6) & 4095);
        const int z = (int)(i >> 18);
        const int b = z >> 4, h = z & 15;
        const float v = qkv[((long long)(b * 4096 + n)) * 3072 + 2048 + h * 64 + e];
        vt[((long long)z * 64 + e) * 4096 + n] = f2bf(v);
    }
}

// ---------------------------------------------------------------------------
extern "C" void kernel_launch(void* const* d_in, const int* in_sizes, int n_in,
                              void* d_out, int out_size, void* d_ws, size_t ws_size,
                              hipStream_t stream) {
    const float* x      = (const float*)d_in[0];   // (4,4096,1024)
    const float* w_qkv  = (const float*)d_in[1];   // (3072,1024)
    const float* w_proj = (const float*)d_in[2];   // (1024,1024)
    const float* b_proj = (const float*)d_in[3];   // (1024,)

    constexpr long long M  = 16384;   // B*N
    constexpr long long C  = 1024;
    constexpr long long C3 = 3072;

    // workspace layout (bytes)
    char* ws = (char*)d_ws;
    float*          qkv    = (float*)ws;                     ws += M * C3 * 4;   // 192MB
    unsigned short* xb     = (unsigned short*)ws;            ws += M * C * 2;    // 32MB
    unsigned short* wqkvb  = (unsigned short*)ws;            ws += C3 * C * 2;   // 6MB
    unsigned short* wprojb = (unsigned short*)ws;            ws += C * C * 2;    // 2MB
    unsigned short* qb     = (unsigned short*)ws;            ws += M * C * 2;    // 32MB
    unsigned short* kt     = (unsigned short*)ws;            ws += 64LL*64*4096*2; // 32MB
    unsigned short* vt     = (unsigned short*)ws;            ws += 64LL*64*4096*2; // 32MB
    unsigned short* ctxT   = (unsigned short*)ws;            ws += 64LL*64*64*2;   // 512KB
    unsigned short* attnb  = (unsigned short*)ws;            /* 32MB */

    // 1) bf16 conversions
    f32_to_bf16_kernel<<<4096, 256, 0, stream>>>(x,      xb,     M * C);
    f32_to_bf16_kernel<<<2048, 256, 0, stream>>>(w_qkv,  wqkvb,  C3 * C);
    f32_to_bf16_kernel<<<1024, 256, 0, stream>>>(w_proj, wprojb, C * C);

    // 2) qkv = x @ w_qkv^T : M=16384, N=3072, K=1024 (fp32 out, LDS-staged)
    gemm_nt_bf16<128, 256, 2, 4, false, false, true>
        <<<dim3(C3 / 256, M / 128, 1), 256, 0, stream>>>(
            xb, wqkvb, qkv, nullptr, (int)C, C, C, C3,
            1, 0, 0, 0, 0, 0, 0);

    // 3) softmax(q) over D -> bf16 qb [16384 x 1024]
    softmax_q_kernel<<<(M * 16) / 8, 256, 0, stream>>>(qkv, qb);

    // 4) softmax(k) over N -> bf16 kT [(bh)][d][n]
    softmax_k_kernel<<<64, 256, 0, stream>>>(qkv, kt);

    // 5) v -> bf16 vT [(bh)][e][n]
    transpose_v_kernel<<<16384, 256, 0, stream>>>(qkv, vt);

    // 6) ctxT[e][d] = sum_n vT[e][n]*kT[d][n] : batched 64, M=N=64, K=4096 (bf16 out)
    gemm_nt_bf16<64, 64, 4, 2, true, false, false>
        <<<dim3(1, 1, 64), 256, 0, stream>>>(
            vt, kt, ctxT, nullptr, 4096, 4096, 4096, 64,
            16,
            16LL * 64 * 4096, 64LL * 4096,    // A offsets (z*64*4096)
            16LL * 64 * 4096, 64LL * 4096,    // B offsets
            16LL * 4096,      4096LL);        // C offsets (z*64*64)

    // 7) attn[n][e] = sum_d q[n][d]*ctxT[e][d] : batched 64, M=4096, N=64, K=64
    //    written into attnb[(b*4096+n)*1024 + h*64 + e] (bf16)
    gemm_nt_bf16<64, 64, 4, 2, true, false, false>
        <<<dim3(1, 4096 / 64, 64), 256, 0, stream>>>(
            qb, ctxT, attnb, nullptr, 64, C, 64, C,
            16,
            4096LL * 1024, 64LL,              // A: b*4096 rows, col h*64
            16LL * 64 * 64, 64LL * 64,        // B: z*4096 elems
            4096LL * 1024, 64LL);             // C: same split as A

    // 8) out = attn @ w_proj^T + b_proj : M=16384, N=1024, K=1024 (fp32 out, LDS-staged)
    gemm_nt_bf16<128, 256, 2, 4, false, true, true>
        <<<dim3(C / 256, M / 128, 1), 256, 0, stream>>>(
            attnb, wprojb, (float*)d_out, b_proj, (int)C, C, C, C,
            1, 0, 0, 0, 0, 0, 0);
}
